// MultiHeadAttention_32495722561673
// MI455X (gfx1250) — compile-verified, matrix operand order checked by reading, not run on hardware
//
#include <hip/hip_runtime.h>
#include <hip/hip_bf16.h>

// ---------------------------------------------------------------------------
// MI455X (gfx1250) multi-head attention with relative position bias.
// All five matmul stages run on v_wmma_f32_16x16x32_bf16 (wave32 WMMA).
// ---------------------------------------------------------------------------

typedef __attribute__((ext_vector_type(16))) __bf16 v16bf;
typedef __attribute__((ext_vector_type(8)))  __bf16 v8bf;
typedef __attribute__((ext_vector_type(8)))  float  v8f;

#define D_MODEL 1024
#define N_HEADS 16
#define D_K     64
#define SEQ     1024
#define BATCH   4
#define NREL    33   // 2*MAX_REL+1

// ---- WMMA wrapper ----------------------------------------------------------
__device__ __forceinline__ v8f wmma_bf16(v16bf a, v16bf b, v8f c) {
  // (neg_a, A, neg_b, B, c_mod, C, reuse_a, reuse_b)
  return __builtin_amdgcn_wmma_f32_16x16x32_bf16(false, a, false, b,
                                                 (short)0, c, false, false);
}

// ---- A-operand tile loaders (16x32, row-major source) ----------------------
// ISA layout (16-bit A 16x32): lanes 0-15 hold K=0..7 (slots 0-7) and K=16..23
// (slots 8-15); lanes 16-31 hold K=8..15 and K=24..31; row M = lane % 16.
__device__ __forceinline__ v16bf loadA_bf16(const __bf16* base, int ld, int lane) {
  int m  = lane & 15;
  int hi = (lane >> 4) & 1;
  const __bf16* p = base + (size_t)m * ld + hi * 8;
  union { v8bf h[2]; v16bf v; } u;
  u.h[0] = *(const v8bf*)(p);
  u.h[1] = *(const v8bf*)(p + 16);
  return u.v;
}

__device__ __forceinline__ v16bf loadA_f32(const float* base, int ld, int lane) {
  int m  = lane & 15;
  int hi = (lane >> 4) & 1;
  const float* p = base + (size_t)m * ld + hi * 8;
  v8f lo = *(const v8f*)(p);
  v8f up = *(const v8f*)(p + 16);
  v16bf v;
#pragma unroll
  for (int i = 0; i < 8; ++i) { v[i] = (__bf16)lo[i]; v[8 + i] = (__bf16)up[i]; }
  return v;
}

// ---- fp32 -> bf16 convert --------------------------------------------------
__global__ void cvt_f32_bf16(const float* __restrict__ in,
                             __bf16* __restrict__ out, int n) {
  int i = blockIdx.x * blockDim.x + threadIdx.x;
  if (i < n) out[i] = (__bf16)in[i];
}

// ---- Generic WMMA GEMM: C[M,N] = A[M,K] @ W[N,K]^T + bias ------------------
// Block = 128 threads = 4 waves (2x2), wave tile 32x64, block tile 64x128.
// MODE 0: write bf16 head-split q/k [B,H,S,dk]
// MODE 2: write bf16 transposed   vT [B,H,dk,S]
// MODE 3: write fp32 row-major    out [M,N]
template <int AISF32, int MODE>
__global__ __launch_bounds__(128)
void gemm_wmma(const void* __restrict__ A,
               const __bf16* __restrict__ Wb, const float* __restrict__ bias,
               void* __restrict__ out, int M, int N, int K) {
  const int lane = threadIdx.x & 31;
  const int wave = threadIdx.x >> 5;
  const int wm = blockIdx.x * 64 + (wave >> 1) * 32;
  const int wn = blockIdx.y * 128 + (wave & 1) * 64;

  v8f z = {0.f, 0.f, 0.f, 0.f, 0.f, 0.f, 0.f, 0.f};
  v8f c[2][4];
#pragma unroll
  for (int i = 0; i < 2; ++i)
#pragma unroll
    for (int j = 0; j < 4; ++j) c[i][j] = z;

  const float*  Af = (const float*)A;
  const __bf16* Ab = (const __bf16*)A;

  for (int kk = 0; kk < K; kk += 32) {
    v16bf a[2], b[4];
#pragma unroll
    for (int i = 0; i < 2; ++i) {
      if (AISF32) {
        const float* p = Af + (size_t)(wm + 16 * i) * K + kk;
        a[i] = loadA_f32(p, K, lane);
        __builtin_prefetch(p + 32, 0, 1);          // next K-step
      } else {
        const __bf16* p = Ab + (size_t)(wm + 16 * i) * K + kk;
        a[i] = loadA_bf16(p, K, lane);
        __builtin_prefetch(p + 32, 0, 1);
      }
    }
#pragma unroll
    for (int j = 0; j < 4; ++j) {
      const __bf16* p = Wb + (size_t)(wn + 16 * j) * K + kk;  // B^T = W rows
      b[j] = loadA_bf16(p, K, lane);
      __builtin_prefetch(p + 32, 0, 1);
    }
#pragma unroll
    for (int i = 0; i < 2; ++i)
#pragma unroll
      for (int j = 0; j < 4; ++j) c[i][j] = wmma_bf16(a[i], b[j], c[i][j]);
  }

  const int hi = lane >> 4;
  const int nl = lane & 15;
#pragma unroll
  for (int i = 0; i < 2; ++i) {
#pragma unroll
    for (int j = 0; j < 4; ++j) {
      int col = wn + j * 16 + nl;
      float bv = bias[col];
#pragma unroll
      for (int r = 0; r < 8; ++r) {
        int row = wm + i * 16 + r + 8 * hi;  // C layout: M = vgpr + 8*(lane/16)
        float val = c[i][j][r] + bv;
        if (MODE == 3) {
          ((float*)out)[(size_t)row * N + col] = val;
        } else {
          int bb = row >> 10, s = row & (SEQ - 1);
          int h  = col >> 6,  d = col & (D_K - 1);
          __bf16* o = (__bf16*)out;
          if (MODE == 2)
            o[(((size_t)(bb * N_HEADS + h)) * D_K + d) * SEQ + s] = (__bf16)val;  // vT
          else
            o[(((size_t)(bb * N_HEADS + h)) * SEQ + s) * D_K + d] = (__bf16)val;  // q/k
        }
      }
    }
  }
}

// ---- relp[bh,s,j] = sum_d q[bh,s,d] * rel_table[j,d] -----------------------
__global__ void relp_kernel(const __bf16* __restrict__ qh,
                            const float* __restrict__ rel_table,
                            float* __restrict__ relp, int total) {
  int idx = blockIdx.x * blockDim.x + threadIdx.x;
  if (idx >= total) return;
  int row = idx / NREL;   // bh*S + s
  int j   = idx - row * NREL;
  const __bf16* q = qh + (size_t)row * D_K;
  const float*  t = rel_table + (size_t)j * D_K;
  float acc = 0.f;
#pragma unroll 8
  for (int d = 0; d < D_K; ++d) acc += (float)q[d] * t[d];
  relp[idx] = acc;
}

// ---- Flash attention, one wave per 16-row l-tile per (b,h) -----------------
// S^T = K @ Q^T via WMMA => C layout has lane = l, VGPR slot = r, which is
// exactly the A-operand layout of P for O = P @ V. No transposes needed.
// Relative bias: gather only on the +-16 diagonal band; elsewhere the clamp
// saturates and the bias is a per-lane preloaded constant (wave-uniform branch).
__global__ __launch_bounds__(128)
void attn_kernel(const __bf16* __restrict__ qh, const __bf16* __restrict__ kh,
                 const __bf16* __restrict__ vT, const float* __restrict__ relp,
                 __bf16* __restrict__ ctx) {
  const int lane = threadIdx.x & 31;
  const int wave = threadIdx.x >> 5;
  const int tile = blockIdx.x * 4 + wave;      // 0 .. B*H*(S/16)-1
  const int bh = tile >> 6;                    // S/16 == 64 tiles per (b,h)
  const int l0 = (tile & 63) * 16;
  const int hi = lane >> 4;

  // Q tile as B-operand (B^T = Q rows in A layout), reused all iterations
  const __bf16* qbase = qh + ((size_t)bh * SEQ + l0) * D_K;
  const v16bf qB0 = loadA_bf16(qbase, D_K, lane);
  const v16bf qB1 = loadA_bf16(qbase + 32, D_K, lane);

  const int l     = lane & 15;
  const int lglob = l0 + l;
  const float* rp = relp + ((size_t)bh * SEQ + lglob) * NREL;
  const float rpLo = rp[0];    // bias for r - l <= -16
  const float rpHi = rp[32];   // bias for r - l >= +16

  float m_run = -1e30f, s_run = 0.f;
  v8f z = {0.f, 0.f, 0.f, 0.f, 0.f, 0.f, 0.f, 0.f};
  v8f o[4];
#pragma unroll
  for (int cidx = 0; cidx < 4; ++cidx) o[cidx] = z;
  const float scale = 0.125f;  // 1/sqrt(64)

  for (int r0 = 0; r0 < SEQ; r0 += 32) {
    // scores^T tiles: t0 covers r0..r0+15, t1 covers r0+16..r0+31
    const __bf16* kb = kh + ((size_t)bh * SEQ + r0) * D_K;
    v16bf kA0 = loadA_bf16(kb, D_K, lane);
    v16bf kA1 = loadA_bf16(kb + 32, D_K, lane);
    v16bf kA2 = loadA_bf16(kb + 16 * D_K, D_K, lane);
    v16bf kA3 = loadA_bf16(kb + 16 * D_K + 32, D_K, lane);
    __builtin_prefetch(kb + 32 * D_K, 0, 1);   // next r-block of K
    v8f t0 = wmma_bf16(kA1, qB1, wmma_bf16(kA0, qB0, z));
    v8f t1 = wmma_bf16(kA3, qB1, wmma_bf16(kA2, qB0, z));

    // scale + relative-position bias, tile row-max
    float p[16];
    float mt = -1e30f;
    // wave-uniform: does this 32-wide r-block touch the +-16 band of any l?
    if (r0 + 31 <= l0 - 16 || r0 >= l0 + 31) {
      const float bconst = (r0 >= l0 + 31) ? rpHi : rpLo;
#pragma unroll
      for (int r = 0; r < 8; ++r) {
        p[r]     = t0[r] * scale + bconst;
        p[8 + r] = t1[r] * scale + bconst;
        float mm = p[r] > p[8 + r] ? p[r] : p[8 + r];
        mt = mt > mm ? mt : mm;
      }
    } else {
#pragma unroll
      for (int r = 0; r < 8; ++r) {
        int rl  = r + 8 * hi;
        int rg0 = r0 + rl;
        int rg1 = r0 + 16 + rl;
        int d0 = rg0 - lglob; if (d0 < -16) d0 = -16; if (d0 > 16) d0 = 16;
        int d1 = rg1 - lglob; if (d1 < -16) d1 = -16; if (d1 > 16) d1 = 16;
        p[r]     = t0[r] * scale + rp[d0 + 16];
        p[8 + r] = t1[r] * scale + rp[d1 + 16];
        float mm = p[r] > p[8 + r] ? p[r] : p[8 + r];
        mt = mt > mm ? mt : mm;
      }
    }
    mt = fmaxf(mt, __shfl_xor(mt, 16, 32));   // halves hold same l
    float m_new = fmaxf(m_run, mt);
    float corr  = __expf(m_run - m_new);

    float sum = 0.f;
    v16bf pA;                                  // P as A-operand (16l x 32r)
#pragma unroll
    for (int r = 0; r < 16; ++r) {
      float e = __expf(p[r] - m_new);
      sum += e;
      pA[r] = (__bf16)e;
    }
    sum += __shfl_xor(sum, 16, 32);
    s_run = s_run * corr + sum;
    m_run = m_new;

    // broadcast per-row rescale factors into the O accumulator layout
    float sc[8];
#pragma unroll
    for (int i = 0; i < 8; ++i) sc[i] = __shfl(corr, i + 8 * hi, 32);

#pragma unroll
    for (int cidx = 0; cidx < 4; ++cidx) {
      // B^T = V^T chunk rows (A layout) straight from transposed v storage
      v16bf vA = loadA_bf16(vT + ((size_t)bh * D_K + cidx * 16) * SEQ + r0,
                            SEQ, lane);
      v8f oc = o[cidx];
#pragma unroll
      for (int i = 0; i < 8; ++i) oc[i] *= sc[i];
      o[cidx] = wmma_bf16(pA, vA, oc);
    }
  }

  // normalize and write ctx[b, s, h*64 + d] in bf16
  float inv[8];
#pragma unroll
  for (int i = 0; i < 8; ++i) inv[i] = 1.0f / __shfl(s_run, i + 8 * hi, 32);
  const int b = bh >> 4, h = bh & 15;
  const int dl = lane & 15;
#pragma unroll
  for (int cidx = 0; cidx < 4; ++cidx) {
#pragma unroll
    for (int i = 0; i < 8; ++i) {
      int row  = l0 + i + 8 * hi;
      int dcol = h * D_K + cidx * 16 + dl;
      ctx[((size_t)b * SEQ + row) * D_MODEL + dcol] = (__bf16)(o[cidx][i] * inv[i]);
    }
  }
}

// ---------------------------------------------------------------------------
extern "C" void kernel_launch(void* const* d_in, const int* in_sizes, int n_in,
                              void* d_out, int out_size, void* d_ws, size_t ws_size,
                              hipStream_t stream) {
  const float* Q   = (const float*)d_in[0];
  const float* Kin = (const float*)d_in[1];
  const float* V   = (const float*)d_in[2];
  const float* Wq  = (const float*)d_in[3];
  const float* bq  = (const float*)d_in[4];
  const float* Wk  = (const float*)d_in[5];
  const float* bk  = (const float*)d_in[6];
  const float* Wv  = (const float*)d_in[7];
  const float* bv  = (const float*)d_in[8];
  const float* Wo  = (const float*)d_in[9];
  const float* bo  = (const float*)d_in[10];
  const float* rel = (const float*)d_in[11];

  const int M = BATCH * SEQ;            // 4096
  const int N = D_MODEL, K = D_MODEL;   // 1024
  const size_t MB = 1024 * 1024;

  char* ws = (char*)d_ws;
  __bf16* Wqb = (__bf16*)(ws + 0 * MB);
  __bf16* Wkb = (__bf16*)(ws + 2 * MB);
  __bf16* Wvb = (__bf16*)(ws + 4 * MB);
  __bf16* Wob = (__bf16*)(ws + 6 * MB);
  __bf16* qhp = (__bf16*)(ws + 8 * MB);    // [B,H,S,dk] bf16  (8 MB)
  __bf16* khp = (__bf16*)(ws + 16 * MB);   // [B,H,S,dk] bf16  (8 MB)
  __bf16* vTp = (__bf16*)(ws + 24 * MB);   // [B,H,dk,S] bf16  (8 MB)
  __bf16* ctx = (__bf16*)(ws + 32 * MB);   // [B,S,D]    bf16  (8 MB)
  float*  rlp = (float*) (ws + 40 * MB);   // [B*H,S,33] fp32  (~8.7 MB)

  // weights -> bf16
  const int nw = D_MODEL * D_MODEL;
  cvt_f32_bf16<<<(nw + 255) / 256, 256, 0, stream>>>(Wq, Wqb, nw);
  cvt_f32_bf16<<<(nw + 255) / 256, 256, 0, stream>>>(Wk, Wkb, nw);
  cvt_f32_bf16<<<(nw + 255) / 256, 256, 0, stream>>>(Wv, Wvb, nw);
  cvt_f32_bf16<<<(nw + 255) / 256, 256, 0, stream>>>(Wo, Wob, nw);

  dim3 gg(M / 64, N / 128);
  // Q/K/V projections (A = fp32 input, converted in-register)
  gemm_wmma<1, 0><<<gg, 128, 0, stream>>>(Q,   Wqb, bq, qhp, M, N, K);
  gemm_wmma<1, 0><<<gg, 128, 0, stream>>>(Kin, Wkb, bk, khp, M, N, K);
  gemm_wmma<1, 2><<<gg, 128, 0, stream>>>(V,   Wvb, bv, vTp, M, N, K);

  // relp[bh,s,j] = q . rel_table[j]
  const int totalRelp = BATCH * N_HEADS * SEQ * NREL;
  relp_kernel<<<(totalRelp + 255) / 256, 256, 0, stream>>>(qhp, rel, rlp, totalRelp);

  // flash attention: B*H*(S/16) = 4096 waves, 4 waves per block
  attn_kernel<<<(BATCH * N_HEADS * (SEQ / 16)) / 4, 128, 0, stream>>>(
      qhp, khp, vTp, rlp, ctx);

  // output projection -> fp32 d_out
  gemm_wmma<0, 3><<<gg, 128, 0, stream>>>(ctx, Wob, bo, (float*)d_out, M, N, K);
}
